// GINe_30227979829590
// MI455X (gfx1250) — matrix operand order repaired and checked by reading, not compiled, not run on hardware
//
#include <hip/hip_runtime.h>
#include <hip/hip_bf16.h>

// ---------------------------------------------------------------------------
// GINE forward for MI455X (gfx1250, wave32, WMMA).
// Memory-bound (~4GB traffic vs 84 GF): fp32 in HBM, f16 in LDS for the matrix
// cores (v_wmma_f32_16x16x32_f16, fp32 accumulate). Edge kernels are persistent
// tile-loops (weights staged once per block), staging is fully vectorized
// (global_load_b128 + packed 8B ds stores), next-tile global_prefetch.
// ---------------------------------------------------------------------------

typedef __attribute__((ext_vector_type(16))) _Float16 v16h;
typedef __attribute__((ext_vector_type(8)))  _Float16 v8h;
typedef __attribute__((ext_vector_type(4)))  _Float16 v4h;
typedef __attribute__((ext_vector_type(8)))  float    v8f;

#define GINE_N      50000
#define GINE_E      1000000
#define GINE_H      64
#define GINE_BNEPS  1e-5f
#define GINE_GINEPS 0.0f
#define GINE_GRID   2048   // persistent blocks for edge-tile loops

#define ATOMIC_ADD_F32(p, v) unsafeAtomicAdd((p), (v))

__device__ __forceinline__ v4h cvt4(float4 f) {
  v4h hv = {(_Float16)f.x, (_Float16)f.y, (_Float16)f.z, (_Float16)f.w};
  return hv;
}

// ---- WMMA fragment loaders (layouts per CDNA5 ISA 7.12.2) ------------------
// A (16x32 f16): lanes 0-15 row M=lane, elems 0..7 -> K = kc+8*hf+e,
//                elems 8..15 -> K = kc+16+8*hf+(e-8). Contiguous 8-half runs.
__device__ __forceinline__ v16h load_a_frag(const _Float16* actRow, int kc, int hf) {
  const _Float16* p = actRow + kc + (hf << 3);
  v8h lo = *(const v8h*)(p);
  v8h hi = *(const v8h*)(p + 16);
  v16h a;
#pragma unroll
  for (int e = 0; e < 8; ++e) { a[e] = lo[e]; a[8 + e] = hi[e]; }
  return a;
}

// B (32x16 f16), weights staged TRANSPOSED in LDS (col-major: wT[col*K + k]):
// lane column N = lane&15, elems 0..7 -> K = kc+8*hf+e, elems 8..15 -> K = kc+16+8*hf+(e-8).
__device__ __forceinline__ v16h load_b_frag(const _Float16* wCol, int kc, int hf) {
  const _Float16* p = wCol + kc + (hf << 3);
  v8h lo = *(const v8h*)(p);
  v8h hi = *(const v8h*)(p + 16);
  v16h b;
#pragma unroll
  for (int e = 0; e < 8; ++e) { b[e] = lo[e]; b[8 + e] = hi[e]; }
  return b;
}

// One wave's 16x64 tile: act (16 rows, stride K, f16 LDS) x wT (64 cols of
// length K, f16 LDS). acc[c] is the 16x16 f32 accumulator for cols c*16..+15.
template <int K>
__device__ __forceinline__ void wmma_tile_16x64(const _Float16* act,
                                                const _Float16* wT, v8f acc[4]) {
  const int lane = threadIdx.x & 31;
  const int hf   = lane >> 4;
  const int l16  = lane & 15;
  const _Float16* actRow = act + l16 * K;
#pragma unroll
  for (int kc = 0; kc < K; kc += 32) {
    v16h a = load_a_frag(actRow, kc, hf);
#pragma unroll
    for (int c = 0; c < 4; ++c) {
      v16h b = load_b_frag(wT + (c * 16 + l16) * K, kc, hf);
      acc[c] = __builtin_amdgcn_wmma_f32_16x16x32_f16(
          false, a, false, b, (short)0, acc[c], false, false);
    }
  }
}

// ---- Kernel 1: embedding GEMM (tile loop), KIN in {32,16} padded to 32 -----
// out[M,64] = in[M,KIN] @ W[KIN,64] + b
template <int KIN>
__global__ void k_embed_gemm(const float* __restrict__ in, const float* __restrict__ W,
                             const float* __restrict__ bias, float* __restrict__ out,
                             int M, int nTiles) {
  __shared__ _Float16 sAct[64 * 32];
  __shared__ _Float16 sWT[64 * 32];
  const int tid = threadIdx.x;
  for (int idx = tid; idx < 64 * 32; idx += 128) {
    int col = idx >> 5, k = idx & 31;
    sWT[col * 32 + k] = (k < KIN) ? (_Float16)W[k * 64 + col] : (_Float16)0.f;
  }
  const int wave = tid >> 5, lane = tid & 31, hf = lane >> 4, l16 = lane & 15;
  float bv[4];
#pragma unroll
  for (int c = 0; c < 4; ++c) bv[c] = bias[c * 16 + l16];
  const float4* in4 = (const float4*)in;

  for (int t = blockIdx.x; t < nTiles; t += gridDim.x) {
    const int m0 = t * 64;
    __syncthreads();
    // stage 64 x 32 (zero-padded) activations: vector loads + packed 8B stores
    for (int idx = tid; idx < 64 * 8; idx += 128) {
      int r = idx >> 3, q = idx & 7;
      int row = m0 + r;
      float4 f = make_float4(0.f, 0.f, 0.f, 0.f);
      if (row < M && q * 4 < KIN) f = in4[row * (KIN / 4) + q];
      *(v4h*)(&sAct[r * 32 + q * 4]) = cvt4(f);
    }
    {  // prefetch next tile's input stream
      int tn = t + gridDim.x;
      if (tn < nTiles && tid * 32 < 64 * KIN)
        __builtin_prefetch(in + (size_t)tn * 64 * KIN + tid * 32, 0, 0);
    }
    __syncthreads();
    v8f acc[4] = {};
    wmma_tile_16x64<32>(sAct + (wave * 16) * 32, sWT, acc);
    if (m0 + 64 <= M) {  // full tile: straight-line stores
#pragma unroll
      for (int c = 0; c < 4; ++c) {
        int gcol = c * 16 + l16;
#pragma unroll
        for (int r = 0; r < 8; ++r) {
          int grow = m0 + wave * 16 + r + hf * 8;
          out[grow * 64 + gcol] = acc[c][r] + bv[c];
        }
      }
    } else {
#pragma unroll
      for (int c = 0; c < 4; ++c) {
        int gcol = c * 16 + l16;
#pragma unroll
        for (int r = 0; r < 8; ++r) {
          int grow = m0 + wave * 16 + r + hf * 8;
          if (grow < M) out[grow * 64 + gcol] = acc[c][r] + bv[c];
        }
      }
    }
  }
}

// ---- Kernel 2: fused GINE message + segment-sum (tile loop) ----------------
// t = ea @ elin_W + elin_b ; msg = relu(h[src] + t) ; agg[dst] += msg
__global__ void k_msg_agg(const float* __restrict__ ea, const float* __restrict__ W,
                          const float* __restrict__ bias, const float* __restrict__ h,
                          const int* __restrict__ src, const int* __restrict__ dst,
                          float* __restrict__ agg, int nTiles) {
  __shared__ _Float16 sAct[64 * 64];
  __shared__ _Float16 sWT[64 * 64];
  const int tid = threadIdx.x;
  for (int idx = tid; idx < 64 * 64; idx += 128) {
    int col = idx >> 6, k = idx & 63;
    sWT[col * 64 + k] = (_Float16)W[k * 64 + col];
  }
  const int wave = tid >> 5, lane = tid & 31, hf = lane >> 4, l16 = lane & 15;
  float bv[4];
#pragma unroll
  for (int c = 0; c < 4; ++c) bv[c] = bias[c * 16 + l16];

  for (int t = blockIdx.x; t < nTiles; t += gridDim.x) {
    const int e0 = t * 64;
    __syncthreads();
    // stage 64x64 ea tile: 1024 b128 loads -> packed 8B ds stores
    const float4* ea4 = (const float4*)(ea + (size_t)e0 * 64);
    for (int idx = tid; idx < 1024; idx += 128)
      *(v4h*)(&sAct[idx * 4]) = cvt4(ea4[idx]);
    {  // prefetch next ea tile (16KB = 128 x 128B lines)
      int tn = t + gridDim.x;
      if (tn < nTiles)
        __builtin_prefetch(ea + (size_t)tn * 4096 + tid * 32, 0, 0);
    }
    __syncthreads();
    v8f acc[4] = {};
    wmma_tile_16x64<64>(sAct + (wave * 16) * 64, sWT, acc);
#pragma unroll
    for (int c = 0; c < 4; ++c) {
      int gcol = c * 16 + l16;
#pragma unroll
      for (int r = 0; r < 8; ++r) {
        int erow = e0 + wave * 16 + r + hf * 8;
        int s = src[erow], d = dst[erow];
        float m = h[s * 64 + gcol] + acc[c][r] + bv[c];
        m = m > 0.f ? m : 0.f;
        ATOMIC_ADD_F32(&agg[d * 64 + gcol], m);
      }
    }
  }
}

// ---- Kernel 3: fused node MLP (two GEMMs) + BN partial stats ---------------
__global__ void k_node_mlp(const float* __restrict__ h, const float* __restrict__ agg,
                           const float* __restrict__ W1, const float* __restrict__ b1,
                           const float* __restrict__ W2, const float* __restrict__ b2,
                           float* __restrict__ out, float* __restrict__ bn_sum,
                           float* __restrict__ bn_sumsq, int M) {
  __shared__ _Float16 sU[64 * 64];
  __shared__ _Float16 sW1T[64 * 64];
  __shared__ _Float16 sW2T[64 * 64];
  __shared__ _Float16 sT1[64 * 64];
  const int tid = threadIdx.x;
  const int m0  = blockIdx.x * 64;
  const bool full = (m0 + 64 <= M);
  for (int idx = tid; idx < 64 * 64; idx += 128) {
    int col = idx >> 6, k = idx & 63;
    sW1T[col * 64 + k] = (_Float16)W1[k * 64 + col];
    sW2T[col * 64 + k] = (_Float16)W2[k * 64 + col];
  }
  if (full) {  // vectorized u = (1+eps)*h + agg
    const float4* h4 = (const float4*)(h + (size_t)m0 * 64);
    const float4* a4 = (const float4*)(agg + (size_t)m0 * 64);
    for (int idx = tid; idx < 1024; idx += 128) {
      float4 hv = h4[idx], av = a4[idx];
      float4 u = make_float4((1.0f + GINE_GINEPS) * hv.x + av.x,
                             (1.0f + GINE_GINEPS) * hv.y + av.y,
                             (1.0f + GINE_GINEPS) * hv.z + av.z,
                             (1.0f + GINE_GINEPS) * hv.w + av.w);
      *(v4h*)(&sU[idx * 4]) = cvt4(u);
    }
  } else {
    for (int idx = tid; idx < 64 * 64; idx += 128) {
      int r = idx >> 6, k = idx & 63;
      int row = m0 + r;
      float v = (row < M)
                    ? ((1.0f + GINE_GINEPS) * h[row * 64 + k] + agg[row * 64 + k])
                    : 0.f;
      sU[idx] = (_Float16)v;
    }
  }
  __syncthreads();
  const int wave = tid >> 5, lane = tid & 31, hf = lane >> 4, l16 = lane & 15;
  {
    v8f acc[4] = {};
    wmma_tile_16x64<64>(sU + (wave * 16) * 64, sW1T, acc);
#pragma unroll
    for (int c = 0; c < 4; ++c) {
      int gcol = c * 16 + l16;
      float bv = b1[gcol];
#pragma unroll
      for (int r = 0; r < 8; ++r) {
        float v = acc[c][r] + bv;
        v = v > 0.f ? v : 0.f;
        sT1[(wave * 16 + r + hf * 8) * 64 + gcol] = (_Float16)v;
      }
    }
  }
  __syncthreads();
  {
    v8f acc[4] = {};
    wmma_tile_16x64<64>(sT1 + (wave * 16) * 64, sW2T, acc);
    if (full) {
#pragma unroll
      for (int c = 0; c < 4; ++c) {
        int gcol = c * 16 + l16;
        float bv = b2[gcol];
        float s = 0.f, sq = 0.f;
#pragma unroll
        for (int r = 0; r < 8; ++r) {
          int grow = m0 + wave * 16 + r + hf * 8;
          float v = acc[c][r] + bv;
          out[grow * 64 + gcol] = v;
          s += v;
          sq += v * v;
        }
        ATOMIC_ADD_F32(&bn_sum[gcol], s);
        ATOMIC_ADD_F32(&bn_sumsq[gcol], sq);
      }
    } else {
#pragma unroll
      for (int c = 0; c < 4; ++c) {
        int gcol = c * 16 + l16;
        float bv = b2[gcol];
        float s = 0.f, sq = 0.f;
#pragma unroll
        for (int r = 0; r < 8; ++r) {
          int grow = m0 + wave * 16 + r + hf * 8;
          if (grow < M) {
            float v = acc[c][r] + bv;
            out[grow * 64 + gcol] = v;
            s += v;
            sq += v * v;
          }
        }
        ATOMIC_ADD_F32(&bn_sum[gcol], s);
        ATOMIC_ADD_F32(&bn_sumsq[gcol], sq);
      }
    }
  }
}

// ---- Kernel 4: fold BN stats into per-feature scale/shift ------------------
__global__ void k_bn_finalize(const float* __restrict__ sum, const float* __restrict__ sumsq,
                              const float* __restrict__ gamma, const float* __restrict__ beta,
                              float* __restrict__ scale, float* __restrict__ shift, float n) {
  int f = threadIdx.x;  // 64 threads
  float mu  = sum[f] / n;
  float var = sumsq[f] / n - mu * mu;
  float sc  = gamma[f] * rsqrtf(var + GINE_BNEPS);
  scale[f] = sc;
  shift[f] = beta[f] - mu * sc;
}

// ---- Kernel 5: h = (h + relu(bn(out))) * 0.5, vectorized -------------------
__global__ void k_h_update(float4* __restrict__ h4, const float4* __restrict__ out4,
                           const float* __restrict__ scale, const float* __restrict__ shift,
                           int n4) {
  int i = blockIdx.x * blockDim.x + threadIdx.x;
  if (i < n4) {
    int c0 = (i & 15) * 4;  // 16 float4 per 64-feature row
    float4 o = out4[i], hh = h4[i];
    float v;
    v = o.x * scale[c0 + 0] + shift[c0 + 0]; v = v > 0.f ? v : 0.f; hh.x = (hh.x + v) * 0.5f;
    v = o.y * scale[c0 + 1] + shift[c0 + 1]; v = v > 0.f ? v : 0.f; hh.y = (hh.y + v) * 0.5f;
    v = o.z * scale[c0 + 2] + shift[c0 + 2]; v = v > 0.f ? v : 0.f; hh.z = (hh.z + v) * 0.5f;
    v = o.w * scale[c0 + 3] + shift[c0 + 3]; v = v > 0.f ? v : 0.f; hh.w = (hh.w + v) * 0.5f;
    h4[i] = hh;
  }
}

// ---- Kernel 6: fused edge update (tile loop) -------------------------------
// e_in = [h[src], h[dst], ea] (K=192) ; hid = relu(e_in@W1+b1) ;
// ea += 0.5 * (hid@W2+b2)
__global__ void k_edge_update(float* __restrict__ ea, const float* __restrict__ h,
                              const int* __restrict__ src, const int* __restrict__ dst,
                              const float* __restrict__ W1, const float* __restrict__ b1,
                              const float* __restrict__ W2, const float* __restrict__ b2,
                              int nTiles) {
  __shared__ _Float16 sIn[64 * 192];   // 24 KB
  __shared__ _Float16 sW1T[64 * 192];  // 24 KB
  __shared__ _Float16 sW2T[64 * 64];   //  8 KB
  __shared__ _Float16 sHid[64 * 64];   //  8 KB  (64 KB total)
  const int tid = threadIdx.x;
  for (int idx = tid; idx < 64 * 192; idx += 128) {
    int col = idx / 192, k = idx % 192;
    sW1T[col * 192 + k] = (_Float16)W1[k * 64 + col];
  }
  for (int idx = tid; idx < 64 * 64; idx += 128) {
    int col = idx >> 6, k = idx & 63;
    sW2T[col * 64 + k] = (_Float16)W2[k * 64 + col];
  }
  const int wave = tid >> 5, lane = tid & 31, hf = lane >> 4, l16 = lane & 15;
  float bv1[4], bv2[4];
#pragma unroll
  for (int c = 0; c < 4; ++c) {
    bv1[c] = b1[c * 16 + l16];
    bv2[c] = b2[c * 16 + l16];
  }
  const float4* hg4 = (const float4*)h;

  for (int t = blockIdx.x; t < nTiles; t += gridDim.x) {
    const int e0 = t * 64;
    __syncthreads();
    // gather h[src] (cols 0..63) and h[dst] (cols 64..127): 16B per vmem op
    for (int idx = tid; idx < 2048; idx += 128) {
      int r = idx >> 5, q = idx & 31;
      int e = e0 + r;
      int node = (q < 16) ? src[e] : dst[e];
      float4 f = hg4[node * 16 + (q & 15)];
      *(v4h*)(&sIn[r * 192 + q * 4]) = cvt4(f);
    }
    // ea tile (cols 128..191): streaming b128 loads
    const float4* ea4 = (const float4*)(ea + (size_t)e0 * 64);
    for (int idx = tid; idx < 1024; idx += 128) {
      int r = idx >> 4, q = idx & 15;
      *(v4h*)(&sIn[r * 192 + 128 + q * 4]) = cvt4(ea4[idx]);
    }
    {  // prefetch next ea tile
      int tn = t + gridDim.x;
      if (tn < nTiles)
        __builtin_prefetch(ea + (size_t)tn * 4096 + tid * 32, 0, 0);
    }
    __syncthreads();
    {
      v8f acc[4] = {};
      wmma_tile_16x64<192>(sIn + (wave * 16) * 192, sW1T, acc);
#pragma unroll
      for (int c = 0; c < 4; ++c) {
        int gcol = c * 16 + l16;
#pragma unroll
        for (int r = 0; r < 8; ++r) {
          float v = acc[c][r] + bv1[c];
          v = v > 0.f ? v : 0.f;
          sHid[(wave * 16 + r + hf * 8) * 64 + gcol] = (_Float16)v;
        }
      }
    }
    __syncthreads();
    {
      v8f acc[4] = {};
      wmma_tile_16x64<64>(sHid + (wave * 16) * 64, sW2T, acc);
#pragma unroll
      for (int c = 0; c < 4; ++c) {
        int gcol = c * 16 + l16;
#pragma unroll
        for (int r = 0; r < 8; ++r) {
          int e = e0 + wave * 16 + r + hf * 8;
          int i = e * 64 + gcol;
          ea[i] = ea[i] + 0.5f * (acc[c][r] + bv2[c]);  // residual read is L2-hot
        }
      }
    }
  }
}

// ---- Kernel 7: zero fill ----------------------------------------------------
__global__ void k_zero4(float4* __restrict__ p, int n4) {
  int i = blockIdx.x * blockDim.x + threadIdx.x;
  if (i < n4) p[i] = make_float4(0.f, 0.f, 0.f, 0.f);
}

// ---------------------------------------------------------------------------
extern "C" void kernel_launch(void* const* d_in, const int* in_sizes, int n_in,
                              void* d_out, int out_size, void* d_ws, size_t ws_size,
                              hipStream_t stream) {
  const float* x        = (const float*)d_in[0];
  const int*   eidx     = (const int*)d_in[1];
  const float* eattr    = (const float*)d_in[2];
  const float* node_W   = (const float*)d_in[3];
  const float* node_b   = (const float*)d_in[4];
  const float* edge_W   = (const float*)d_in[5];
  const float* edge_b   = (const float*)d_in[6];
  const float* elin_W   = (const float*)d_in[7];
  const float* elin_b   = (const float*)d_in[8];
  const float* mlp1_W   = (const float*)d_in[9];
  const float* mlp1_b   = (const float*)d_in[10];
  const float* mlp2_W   = (const float*)d_in[11];
  const float* mlp2_b   = (const float*)d_in[12];
  const float* emlp1_W  = (const float*)d_in[13];
  const float* emlp1_b  = (const float*)d_in[14];
  const float* emlp2_W  = (const float*)d_in[15];
  const float* emlp2_b  = (const float*)d_in[16];
  const float* bn_gamma = (const float*)d_in[17];
  const float* bn_beta  = (const float*)d_in[18];

  const int N = GINE_N, E = GINE_E;
  const int* src = eidx;
  const int* dst = eidx + E;

  // Outputs: h [N*64] then ea [E*64], flat in return order.
  float* h  = (float*)d_out;
  float* ea = (float*)d_out + (size_t)N * 64;

  // Workspace: agg [N*64] | tmp_out [N*64] | bn_sum[64] | bn_sq[64] |
  //            bn_scale[64] | bn_shift[64]   (~25.6 MB)
  float* ws       = (float*)d_ws;
  float* agg      = ws;                 ws += (size_t)N * 64;
  float* tmp_out  = ws;                 ws += (size_t)N * 64;
  float* bn_sum   = ws;                 ws += 64;
  float* bn_sq    = ws;                 ws += 64;
  float* bn_scale = ws;                 ws += 64;
  float* bn_shift = ws;                 ws += 64;

  const dim3 blk(128);                       // 4 waves (wave32)
  const int nTiles = (N + 63) / 64;          // 782
  const int eTiles = E / 64;                 // 15625
  const int nGrid  = (nTiles < GINE_GRID) ? nTiles : GINE_GRID;
  const int eGrid  = GINE_GRID;

  // Embeddings
  k_embed_gemm<32><<<nGrid, blk, 0, stream>>>(x, node_W, node_b, h, N, nTiles);
  k_embed_gemm<16><<<eGrid, blk, 0, stream>>>(eattr, edge_W, edge_b, ea, E, eTiles);

  for (int i = 0; i < 2; ++i) {
    k_zero4<<<(N * 16 + 255) / 256, 256, 0, stream>>>((float4*)agg, N * 16);
    k_zero4<<<1, 32, 0, stream>>>((float4*)bn_sum, 32);  // bn_sum+bn_sq (128 f)

    k_msg_agg<<<eGrid, blk, 0, stream>>>(ea, elin_W + i * 64 * 64,
                                         elin_b + i * 64, h, src, dst, agg,
                                         eTiles);

    k_node_mlp<<<nTiles, blk, 0, stream>>>(h, agg,
                                           mlp1_W + i * 64 * 64, mlp1_b + i * 64,
                                           mlp2_W + i * 64 * 64, mlp2_b + i * 64,
                                           tmp_out, bn_sum, bn_sq, N);

    k_bn_finalize<<<1, 64, 0, stream>>>(bn_sum, bn_sq, bn_gamma + i * 64,
                                        bn_beta + i * 64, bn_scale, bn_shift,
                                        (float)N);

    k_h_update<<<(N * 16 + 255) / 256, 256, 0, stream>>>((float4*)h,
                                                         (const float4*)tmp_out,
                                                         bn_scale, bn_shift,
                                                         N * 16);

    k_edge_update<<<eGrid, blk, 0, stream>>>(ea, h, src, dst,
                                             emlp1_W + i * 192 * 64,
                                             emlp1_b + i * 64,
                                             emlp2_W + i * 64 * 64,
                                             emlp2_b + i * 64, eTiles);
  }
}
